// Layer_49057116455430
// MI455X (gfx1250) — matrix operand order repaired and checked by reading, not compiled, not run on hardware
//
#include <hip/hip_runtime.h>
#include <hip/hip_bf16.h>
#include <stdint.h>

#define S_REAL 1213
#define S_PAD  1216          // 19 * 64
#define NBATCH 8
#define HID 768
#define HEADS 12
#define DHEAD 64
#define INTER_DIM 3072
#define M_ROWS (NBATCH * S_PAD)   // 9728

typedef __attribute__((ext_vector_type(16))) _Float16 v16h;
typedef __attribute__((ext_vector_type(8)))  float    v8f;

// POD 16-byte vector (HIP's uint4 has non-trivial ctors -> unusable in a union)
struct alignas(16) U4 { unsigned int x, y, z, w; };
union F16Frag { v16h h; U4 u[2]; };

static __device__ __forceinline__ v8f wmma_f16(v16h a, v16h b, v8f c) {
  // D = A(16x32 f16) * B(32x16 f16) + C(16x16 f32)
  return __builtin_amdgcn_wmma_f32_16x16x32_f16(false, a, false, b, (short)0, c, false, false);
}

static __device__ __forceinline__ float lead_mask_val(int q, int key) {
  if (q == 0) return (key >= 1 && key < 13) ? -99999.0f : 0.0f;
  if (q <= 12) {
    int st = 13 + 100 * (q - 1);
    return (key >= st && key < st + 100) ? 0.0f : -99999.0f;
  }
  return 0.0f;
}

// ---------------- converts ----------------
__global__ __launch_bounds__(256) void convert_x_kernel(const float* __restrict__ hs,
                                                        _Float16* __restrict__ xh) {
  size_t idx = (size_t)blockIdx.x * 256 + threadIdx.x;
  if (idx >= (size_t)M_ROWS * HID) return;
  size_t r = idx / HID, c = idx % HID;
  int b = (int)(r / S_PAD), t = (int)(r % S_PAD);
  float v = (t < S_REAL) ? hs[((size_t)b * S_REAL + t) * HID + c] : 0.0f;
  xh[idx] = (_Float16)v;
}

// W[k][n] f32 -> Wt[n][k] f16
__global__ __launch_bounds__(256) void convert_w_kernel(const float* __restrict__ W,
                                                        _Float16* __restrict__ Wt,
                                                        int K, int N) {
  size_t idx = (size_t)blockIdx.x * 256 + threadIdx.x;
  if (idx >= (size_t)K * N) return;
  size_t k = idx / N, n = idx % N;
  Wt[n * (size_t)K + k] = (_Float16)W[idx];
}

// ---------------- generic WMMA GEMM ----------------
// C[M_ROWS, N] = A[M_ROWS, K](f16) * Bt[N, K](f16)^T + bias, with per-MODE epilogue.
// MODE 0: Q  -> f16 [bh][s][64], scaled 0.125     MODE 1: K -> f16 [bh][s][64]
// MODE 2: V  -> f16 [bh][64][s] (transposed)      MODE 3: +resid(hidden f32) -> f32
// MODE 4: GELU -> f16                              MODE 5: +resid(f32 [M,HID]) -> f32
template <int MODE>
__global__ __launch_bounds__(128) void gemm_kernel(const _Float16* __restrict__ A,
                                                   const _Float16* __restrict__ Bt,
                                                   const float* __restrict__ bias,
                                                   const float* __restrict__ resid,
                                                   void* __restrict__ outp, int K) {
  __shared__ __align__(16) _Float16 Ash[64][40];
  __shared__ __align__(16) _Float16 Bsh[64][40];

  const int tid = threadIdx.x;
  const int lane = tid & 31, w = tid >> 5;
  const int h = lane >> 4, ln = lane & 15;
  const int wm = (w & 1) * 32, wn = (w >> 1) * 32;
  const long m0 = (long)blockIdx.x * 64;
  const long n0 = (long)blockIdx.y * 64;

  v8f acc[2][2] = {};

  for (int k0 = 0; k0 < K; k0 += 32) {
#pragma unroll
    for (int jj = 0; jj < 2; ++jj) {           // 256 16B chunks, 2 per thread
      int c = tid + 128 * jj;
      int row = c >> 2, c8 = c & 3;
      *reinterpret_cast<U4*>(&Ash[row][c8 * 8]) =
          *reinterpret_cast<const U4*>(A + (size_t)(m0 + row) * K + k0 + c8 * 8);
      *reinterpret_cast<U4*>(&Bsh[row][c8 * 8]) =
          *reinterpret_cast<const U4*>(Bt + (size_t)(n0 + row) * K + k0 + c8 * 8);
    }
    if (k0 + 32 < K) {                         // global_prefetch_b8 of next tiles
      __builtin_prefetch(A + (size_t)(m0 + (tid >> 1)) * K + k0 + 32, 0, 1);
      __builtin_prefetch(Bt + (size_t)(n0 + (tid >> 1)) * K + k0 + 32, 0, 1);
    }
    __syncthreads();

    F16Frag af[2], bf[2];
#pragma unroll
    for (int i = 0; i < 2; ++i) {              // A frag: row=lane&15, K pairs in VGPRs
      int row = wm + i * 16 + ln;
      af[i].u[0] = *reinterpret_cast<const U4*>(&Ash[row][8 * h]);
      af[i].u[1] = *reinterpret_cast<const U4*>(&Ash[row][16 + 8 * h]);
    }
#pragma unroll
    for (int j = 0; j < 2; ++j) {              // B frag: N=lane&15, K pairs in VGPRs
      int row = wn + j * 16 + ln;
      bf[j].u[0] = *reinterpret_cast<const U4*>(&Bsh[row][8 * h]);
      bf[j].u[1] = *reinterpret_cast<const U4*>(&Bsh[row][16 + 8 * h]);
    }
#pragma unroll
    for (int i = 0; i < 2; ++i)
#pragma unroll
      for (int j = 0; j < 2; ++j) acc[i][j] = wmma_f16(af[i].h, bf[j].h, acc[i][j]);
    __syncthreads();
  }

#pragma unroll
  for (int i = 0; i < 2; ++i) {
#pragma unroll
    for (int j = 0; j < 2; ++j) {
#pragma unroll
      for (int v = 0; v < 8; ++v) {
        long r = m0 + wm + i * 16 + v + 8 * h;   // C layout: M = vgpr + 8*half
        long n = n0 + wn + j * 16 + ln;          //           N = lane&15
        int b = (int)(r / S_PAD), t = (int)(r % S_PAD);
        float val = acc[i][j][v] + bias[n];
        if (MODE == 0) {
          _Float16* q = (_Float16*)outp;
          int head = (int)(n >> 6), d = (int)(n & 63);
          q[((size_t)(b * HEADS + head) * S_PAD + t) * DHEAD + d] = (_Float16)(val * 0.125f);
        } else if (MODE == 1) {
          _Float16* kq = (_Float16*)outp;
          int head = (int)(n >> 6), d = (int)(n & 63);
          kq[((size_t)(b * HEADS + head) * S_PAD + t) * DHEAD + d] = (_Float16)val;
        } else if (MODE == 2) {
          _Float16* vt = (_Float16*)outp;
          int head = (int)(n >> 6), d = (int)(n & 63);
          vt[((size_t)(b * HEADS + head) * DHEAD + d) * S_PAD + t] = (_Float16)val;
        } else if (MODE == 3) {
          float* o = (float*)outp;
          float res = (t < S_REAL) ? resid[((size_t)b * S_REAL + t) * HID + n] : 0.0f;
          o[(size_t)r * HID + n] = val + res;
        } else if (MODE == 4) {
          _Float16* o = (_Float16*)outp;
          float g = 0.5f * val * (1.0f + erff(val * 0.70710678118654752f));
          o[(size_t)r * INTER_DIM + n] = (_Float16)g;
        } else {
          float* o = (float*)outp;
          o[(size_t)r * HID + n] = val + resid[(size_t)r * HID + n];
        }
      }
    }
  }
}

// ---------------- flash attention ----------------
// grid = (S_PAD/64, B*HEADS), block = 128 (4 waves, 16 query rows each)
__global__ __launch_bounds__(128) void attn_kernel(const _Float16* __restrict__ qh,
                                                   const _Float16* __restrict__ kh,
                                                   const _Float16* __restrict__ vth,
                                                   const float* __restrict__ amask,
                                                   _Float16* __restrict__ ctxh) {
  __shared__ __align__(16) _Float16 Plds[4][16][40];

  const int tid = threadIdx.x;
  const int lane = tid & 31, w = tid >> 5;
  const int h = lane >> 4, ln = lane & 15;
  const int bh = blockIdx.y;
  const int b = bh / HEADS, head = bh % HEADS;
  const int qbase = blockIdx.x * 64 + w * 16;

  // Q A-fragments (d = 0..31 and 32..63), Q is pre-scaled by 1/sqrt(64)
  const _Float16* qrow = qh + ((size_t)bh * S_PAD + qbase + ln) * DHEAD;
  F16Frag qa0, qa1;
  qa0.u[0] = *reinterpret_cast<const U4*>(qrow + 8 * h);
  qa0.u[1] = *reinterpret_cast<const U4*>(qrow + 16 + 8 * h);
  qa1.u[0] = *reinterpret_cast<const U4*>(qrow + 32 + 8 * h);
  qa1.u[1] = *reinterpret_cast<const U4*>(qrow + 48 + 8 * h);

  v8f oacc[4] = {};
  float mrow[8], lrow[8];
#pragma unroll
  for (int v = 0; v < 8; ++v) { mrow[v] = -3.0e38f; lrow[v] = 0.0f; }

  for (int k0 = 0; k0 < S_PAD; k0 += 32) {
    v8f s[2];
#pragma unroll
    for (int nt = 0; nt < 2; ++nt) {           // scores: Q * K^T, B-frag straight from K rows
      int key = k0 + nt * 16 + ln;
      const _Float16* krow = kh + ((size_t)bh * S_PAD + key) * DHEAD;
      F16Frag kb0, kb1;
      kb0.u[0] = *reinterpret_cast<const U4*>(krow + 8 * h);
      kb0.u[1] = *reinterpret_cast<const U4*>(krow + 16 + 8 * h);
      kb1.u[0] = *reinterpret_cast<const U4*>(krow + 32 + 8 * h);
      kb1.u[1] = *reinterpret_cast<const U4*>(krow + 48 + 8 * h);
      v8f z = {};
      z = wmma_f16(qa0.h, kb0.h, z);
      z = wmma_f16(qa1.h, kb1.h, z);
      s[nt] = z;
    }
    // masks
#pragma unroll
    for (int nt = 0; nt < 2; ++nt) {
      int key = k0 + nt * 16 + ln;
      float am = (key < S_REAL) ? amask[(size_t)b * S_REAL + key] : 0.0f;
#pragma unroll
      for (int v = 0; v < 8; ++v) {
        int q = qbase + v + 8 * h;
        float sc = s[nt][v];
        if (key >= S_REAL) sc = -3.0e38f;            // padded keys never contribute
        else sc += am + lead_mask_val(q, key);
        s[nt][v] = sc;
      }
    }
    // online softmax (row = v + 8h lives in the 16-lane half -> xor<=8 reductions)
#pragma unroll
    for (int v = 0; v < 8; ++v) {
      float mx = fmaxf(s[0][v], s[1][v]);
#pragma unroll
      for (int d = 1; d < 16; d <<= 1) mx = fmaxf(mx, __shfl_xor(mx, d, 32));
      float mnew = fmaxf(mrow[v], mx);
      float p0 = __expf(s[0][v] - mnew);
      float p1 = __expf(s[1][v] - mnew);
      float scale = __expf(mrow[v] - mnew);
      float ps = p0 + p1;
#pragma unroll
      for (int d = 1; d < 16; d <<= 1) ps += __shfl_xor(ps, d, 32);
      lrow[v] = lrow[v] * scale + ps;
      mrow[v] = mnew;
#pragma unroll
      for (int dt = 0; dt < 4; ++dt) oacc[dt][v] *= scale;
      Plds[w][v + 8 * h][ln] = (_Float16)p0;        // stage probs: C-layout -> LDS
      Plds[w][v + 8 * h][16 + ln] = (_Float16)p1;
    }
    __syncthreads();
    F16Frag pa;                                     // reload probs as A-fragment
    pa.u[0] = *reinterpret_cast<const U4*>(&Plds[w][ln][8 * h]);
    pa.u[1] = *reinterpret_cast<const U4*>(&Plds[w][ln][16 + 8 * h]);
#pragma unroll
    for (int dt = 0; dt < 4; ++dt) {                // O += P * V  (Vt: [d][key] -> B-frag)
      const _Float16* vrow = vth + ((size_t)bh * DHEAD + dt * 16 + ln) * S_PAD + k0 + 8 * h;
      F16Frag vb;
      vb.u[0] = *reinterpret_cast<const U4*>(vrow);
      vb.u[1] = *reinterpret_cast<const U4*>(vrow + 16);
      oacc[dt] = wmma_f16(pa.h, vb.h, oacc[dt]);
    }
    __syncthreads();
  }

#pragma unroll
  for (int v = 0; v < 8; ++v) {
    float inv = 1.0f / lrow[v];
    int t = qbase + v + 8 * h;
    size_t base = ((size_t)b * S_PAD + t) * HID + head * DHEAD;
#pragma unroll
    for (int dt = 0; dt < 4; ++dt)
      ctxh[base + dt * 16 + ln] = (_Float16)(oacc[dt][v] * inv);
  }
}

// ---------------- layer norm ----------------
__global__ __launch_bounds__(256) void ln_kernel(const float* __restrict__ in,
                                                 const float* __restrict__ gamma,
                                                 const float* __restrict__ beta,
                                                 float* __restrict__ outf,
                                                 _Float16* __restrict__ outh,
                                                 int final_out) {
  __shared__ float red[256];
  const int r = blockIdx.x, tid = threadIdx.x;
  const int b = r / S_PAD, t = r % S_PAD;

  if (t >= S_REAL) {                 // padded rows: zero (or skip for final output)
    if (!final_out) {
      for (int c = tid; c < HID; c += 256) {
        outf[(size_t)r * HID + c] = 0.0f;
        outh[(size_t)r * HID + c] = (_Float16)0.0f;
      }
    }
    return;
  }

  float x0 = in[(size_t)r * HID + tid];
  float x1 = in[(size_t)r * HID + tid + 256];
  float x2 = in[(size_t)r * HID + tid + 512];
  red[tid] = x0 + x1 + x2;
  __syncthreads();
  for (int st = 128; st > 0; st >>= 1) { if (tid < st) red[tid] += red[tid + st]; __syncthreads(); }
  float mean = red[0] / 768.0f;
  __syncthreads();
  float d0 = x0 - mean, d1 = x1 - mean, d2 = x2 - mean;
  red[tid] = d0 * d0 + d1 * d1 + d2 * d2;
  __syncthreads();
  for (int st = 128; st > 0; st >>= 1) { if (tid < st) red[tid] += red[tid + st]; __syncthreads(); }
  float rstd = rsqrtf(red[0] / 768.0f + 1e-12f);

  float dd[3] = { d0, d1, d2 };
#pragma unroll
  for (int k = 0; k < 3; ++k) {
    int c = tid + 256 * k;
    float y = gamma[c] * dd[k] * rstd + beta[c];
    if (final_out) {
      outf[((size_t)b * S_REAL + t) * HID + c] = y;
    } else {
      outf[(size_t)r * HID + c] = y;
      outh[(size_t)r * HID + c] = (_Float16)y;
    }
  }
}

// ---------------- host ----------------
extern "C" void kernel_launch(void* const* d_in, const int* in_sizes, int n_in,
                              void* d_out, int out_size, void* d_ws, size_t ws_size,
                              hipStream_t stream) {
  (void)in_sizes; (void)n_in; (void)out_size; (void)ws_size;
  const float* hs    = (const float*)d_in[0];
  const float* amask = (const float*)d_in[1];
  const float* Wq = (const float*)d_in[2];  const float* bq = (const float*)d_in[3];
  const float* Wk = (const float*)d_in[4];  const float* bk = (const float*)d_in[5];
  const float* Wv = (const float*)d_in[6];  const float* bv = (const float*)d_in[7];
  const float* Wo = (const float*)d_in[8];  const float* bo = (const float*)d_in[9];
  const float* g1 = (const float*)d_in[10]; const float* be1 = (const float*)d_in[11];
  const float* Wi = (const float*)d_in[12]; const float* bi = (const float*)d_in[13];
  const float* Wd = (const float*)d_in[14]; const float* bd = (const float*)d_in[15];
  const float* g2 = (const float*)d_in[16]; const float* be2 = (const float*)d_in[17];
  float* out = (float*)d_out;

  char* ws = (char*)d_ws;
  size_t off = 0;
  auto take = [&](size_t bytes) {
    char* p = ws + off;
    off += (bytes + 255) & ~(size_t)255;
    return p;
  };
  const size_t SZ_W768  = (size_t)768 * 768 * 2;
  const size_t SZ_W3072 = (size_t)768 * 3072 * 2;
  const size_t SZ_ACT_H = (size_t)M_ROWS * HID * 2;
  const size_t SZ_ACT_F = (size_t)M_ROWS * HID * 4;
  const size_t SZ_INT_H = (size_t)M_ROWS * INTER_DIM * 2;

  _Float16* wtq = (_Float16*)take(SZ_W768);
  _Float16* wtk = (_Float16*)take(SZ_W768);
  _Float16* wtv = (_Float16*)take(SZ_W768);
  _Float16* wto = (_Float16*)take(SZ_W768);
  _Float16* wti = (_Float16*)take(SZ_W3072);
  _Float16* wtd = (_Float16*)take(SZ_W3072);
  _Float16* xh  = (_Float16*)take(SZ_ACT_H);   // reused as ctxh after attention
  _Float16* qh  = (_Float16*)take(SZ_ACT_H);   // reused as attn_h after attention
  _Float16* kh  = (_Float16*)take(SZ_ACT_H);
  _Float16* vth = (_Float16*)take(SZ_ACT_H);
  float*    pre = (float*)take(SZ_ACT_F);      // pre-LN buffer (used twice)
  float*    attnf = (float*)take(SZ_ACT_F);    // attn_out f32 residual
  _Float16* interh = (_Float16*)take(SZ_INT_H);
  _Float16* ctxh  = xh;
  _Float16* attnh = qh;

  // converts
  {
    size_t tot = (size_t)M_ROWS * HID;
    convert_x_kernel<<<dim3((unsigned)((tot + 255) / 256)), dim3(256), 0, stream>>>(hs, xh);
    unsigned bw = (768u * 768u + 255u) / 256u;
    convert_w_kernel<<<dim3(bw), dim3(256), 0, stream>>>(Wq, wtq, 768, 768);
    convert_w_kernel<<<dim3(bw), dim3(256), 0, stream>>>(Wk, wtk, 768, 768);
    convert_w_kernel<<<dim3(bw), dim3(256), 0, stream>>>(Wv, wtv, 768, 768);
    convert_w_kernel<<<dim3(bw), dim3(256), 0, stream>>>(Wo, wto, 768, 768);
    unsigned bw2 = (768u * 3072u + 255u) / 256u;
    convert_w_kernel<<<dim3(bw2), dim3(256), 0, stream>>>(Wi, wti, 768, 3072);
    convert_w_kernel<<<dim3(bw2), dim3(256), 0, stream>>>(Wd, wtd, 3072, 768);
  }

  dim3 blk(128);
  dim3 g768(M_ROWS / 64, HID / 64);        // (152, 12)
  dim3 g3072(M_ROWS / 64, INTER_DIM / 64); // (152, 48)

  gemm_kernel<0><<<g768, blk, 0, stream>>>(xh, wtq, bq, nullptr, qh, 768);
  gemm_kernel<1><<<g768, blk, 0, stream>>>(xh, wtk, bk, nullptr, kh, 768);
  gemm_kernel<2><<<g768, blk, 0, stream>>>(xh, wtv, bv, nullptr, vth, 768);

  attn_kernel<<<dim3(S_PAD / 64, NBATCH * HEADS), blk, 0, stream>>>(qh, kh, vth, amask, ctxh);

  gemm_kernel<3><<<g768, blk, 0, stream>>>(ctxh, wto, bo, hs, pre, 768);
  ln_kernel<<<dim3(M_ROWS), dim3(256), 0, stream>>>(pre, g1, be1, attnf, attnh, 0);

  gemm_kernel<4><<<g3072, blk, 0, stream>>>(attnh, wti, bi, nullptr, interh, 768);
  gemm_kernel<5><<<g768, blk, 0, stream>>>(interh, wtd, bd, attnf, pre, 3072);
  ln_kernel<<<dim3(M_ROWS), dim3(256), 0, stream>>>(pre, g2, be2, out, nullptr, 1);
}
